// CrossModalMoELayer_29592324669899
// MI455X (gfx1250) — matrix operand order repaired and checked by reading, not compiled, not run on hardware
//
#include <hip/hip_runtime.h>

// ---------------------------------------------------------------------------
// CDNA5 (gfx1250) implementation of CrossModalMoELayer.
// All GEMMs run through a bf16 WMMA (v_wmma_f32_16x16x32_bf16) tiled kernel,
// f32 accumulate. Compute-bound workload (~620 GFLOP, AI ~1.5 kFLOP/B).
// Tile staging uses the Tensor Data Mover (tensor_load_to_lds + TENSORcnt)
// when the toolchain exposes it; branchless vector staging otherwise.
// NOTE: all GEMM launches use K % 32 == 0 (no K-tail path compiled).
// ---------------------------------------------------------------------------

typedef __bf16 bf16_t;
typedef __attribute__((ext_vector_type(8)))  bf16_t bf16x8;
typedef __attribute__((ext_vector_type(16))) bf16_t bf16x16;
typedef __attribute__((ext_vector_type(8)))  float  f32x8;
typedef unsigned int v4u __attribute__((ext_vector_type(4)));
typedef int v8i __attribute__((ext_vector_type(8)));
typedef int v4i __attribute__((ext_vector_type(4)));

#define BM 128
#define BN 128
#define BKK 32
#define LDSK 40   // 32 + 8 pad (80B row pitch -> conflict-free frag reads)

#if __has_builtin(__builtin_amdgcn_tensor_load_to_lds) && __has_builtin(__builtin_amdgcn_s_wait_tensorcnt)
#define USE_TDM 1
#else
#define USE_TDM 0
#endif

__device__ __forceinline__ float gelu_tanh(float x) {
  const float c0 = 0.7978845608028654f;
  const float c1 = 0.044715f;
  return 0.5f * x * (1.0f + tanhf(c0 * (x + c1 * x * x * x)));
}

#if USE_TDM
// Issue one TDM 2D-tile load: [rows x BKK bf16] from global (row pitch `ld`
// elements) into LDS at byte offset lds_off, padded to LDSK-element pitch.
// D# layout per CDNA5 ISA ch.8 (group0 128b, group1 256b; groups 2/3 zero).
// This toolchain exposes the 6-arg builtin:
//   (uint32x4 g0, int32x8 g1, int32x4 g2, int32x4 g3, int32x8 gx, i32 cpol)
__device__ __forceinline__ void tdm_load_tile(unsigned lds_off, const bf16_t* gaddr,
                                              int rows_rem, int k_rem, int ld) {
  unsigned long long ga = (unsigned long long)(uintptr_t)gaddr;
  v4u g0;
  g0[0] = 1u;                                          // count=1, user descriptor
  g0[1] = lds_off;                                     // lds_addr (bytes)
  g0[2] = (unsigned)(ga & 0xffffffffu);                // global_addr[31:0]
  g0[3] = (unsigned)((ga >> 32) & 0x1ffffffu) | (2u << 30);  // addr[56:32] | type=2
  unsigned d0 = (unsigned)k_rem;                       // tensor_dim0 (elements)
  unsigned d1 = (unsigned)rows_rem;                    // tensor_dim1 (rows; OOB reads 0)
  v8i g1;
  g1[0] = (int)((1u << 16)      // data_size = 2 bytes
              | (1u << 20)      // pad_enable
              | (3u << 22)      // pad_interval: 16 dwords (= one 64B tile row)
              | (3u << 25));    // pad_amount: 4 dwords (16B -> 80B pitch = LDSK)
  g1[1] = (int)((d0 & 0xffffu) << 16);                               // dim0[15:0]
  g1[2] = (int)(((d0 >> 16) & 0xffffu) | ((d1 & 0xffffu) << 16));    // dim0 hi | dim1 lo
  g1[3] = (int)(((d1 >> 16) & 0xffffu) | ((unsigned)BKK << 16));     // dim1 hi | tile_dim0
  g1[4] = (int)(unsigned)BM;                                         // tile_dim1 | tile_dim2=0
  g1[5] = (int)(unsigned)ld;                                         // dim0_stride lo
  g1[6] = 0;                                                         // stride hi | dim1_stride
  g1[7] = 0;
  v4i z4 = {0, 0, 0, 0};
  v8i z8 = {0, 0, 0, 0, 0, 0, 0, 0};
  __builtin_amdgcn_tensor_load_to_lds(g0, g1, z4, z4, z8, 0);
}
#endif

// EPI: 0 = f32 out (alpha*acc + bias)
//      1 = bf16 out (alpha*acc + bias)
//      2 = bf16 out gelu(acc + bias)
//      3 = f32 accumulate: C += rowscale[m]*(acc + bias)   (rowscale NULL -> 1)
// OPB: 0 = B is [N,K] row-major (compute A*B^T), 1 = B is [K,N] row-major
template <int EPI, int OPB>
__global__ __launch_bounds__(256)
void gemm_bf16_k(const bf16_t* __restrict__ A, const bf16_t* __restrict__ Bm,
                 const float* __restrict__ bias,
                 float* __restrict__ Cf, bf16_t* __restrict__ Cb,
                 int M, int N, int K, int lda, int ldb, int ldc,
                 long sAb, long sAh, long sBb, long sBh, long sCb, long sCh,
                 int nh, float alpha, const float* __restrict__ rs, int rsStride) {
  __shared__ bf16_t ldsbuf[(BM + BN) * LDSK];
  bf16_t* ldsA = ldsbuf;                 // static LDS starts at offset 0
  bf16_t* ldsB = ldsbuf + BM * LDSK;

  const int tid  = threadIdx.x;
  const int lane = tid & 31;
  const int wave = tid >> 5;
  const int wm = wave >> 2;  // 0..1 -> 64 rows each
  const int wn = wave & 3;   // 0..3 -> 32 cols each

  const int m0 = blockIdx.y * BM;
  const int n0 = blockIdx.x * BN;
  const int z  = blockIdx.z;
  const int bb = z / nh;
  const int hh = z - bb * nh;
  const long offA = (long)bb * sAb + (long)hh * sAh;
  const long offB = (long)bb * sBb + (long)hh * sBh;
  const long offC = (long)bb * sCb + (long)hh * sCh;

  f32x8 acc[4][2];
#pragma unroll
  for (int i = 0; i < 4; i++)
#pragma unroll
    for (int j = 0; j < 2; j++)
#pragma unroll
      for (int r = 0; r < 8; r++) acc[i][j][r] = 0.0f;

  const int kbA = (lane >> 4) * 8;    // A frag: K chunks at kbA, kbA+16
  const int kbB = (lane >> 4) * 8;    // B frag: K chunks at kbB, kbB+16
  const int rA  = lane & 15;

  for (int k0 = 0; k0 < K; k0 += BKK) {
#if USE_TDM
    // -- TDM staging: wave 0 DMAs A (and B for the A*B^T case) into LDS --
    if (tid < 32) {
      tdm_load_tile(0u, A + offA + (long)m0 * lda + k0, M - m0, K - k0, lda);
      if (OPB == 0)
        tdm_load_tile((unsigned)(BM * LDSK * 2),
                      Bm + offB + (long)n0 * ldb + k0, N - n0, K - k0, ldb);
    }
#else
    {  // branchless stage A [BM x 32]; K % 32 == 0 so no K guard
      int row = tid >> 1;
      int ck  = (tid & 1) * 16;
      bool ok = (m0 + row) < M;
      long roff = ok ? (long)(m0 + row) * lda : 0;   // clamp to row 0 (always valid)
      uint4 u = *(const uint4*)(A + offA + roff + k0 + ck);
      if (!ok) { u.x = 0u; u.y = 0u; u.z = 0u; u.w = 0u; }   // v_cndmask, no branch
      *(uint4*)&ldsA[row * LDSK + ck] = u;
    }
    if (OPB == 0) {  // branchless stage B [BN x 32] (B is [N,K])
      int row = tid >> 1;
      int ck  = (tid & 1) * 16;
      bool ok = (n0 + row) < N;
      long roff = ok ? (long)(n0 + row) * ldb : 0;
      uint4 u = *(const uint4*)(Bm + offB + roff + k0 + ck);
      if (!ok) { u.x = 0u; u.y = 0u; u.z = 0u; u.w = 0u; }
      *(uint4*)&ldsB[row * LDSK + ck] = u;
    }
#endif
    if (OPB == 1) {  // transpose-stage B (B is [K,N]) -> lds [N][K]; N % 16 == 0
      int kk = tid >> 3;        // 0..31
      int c0 = (tid & 7) * 16;  // 0..112
      bool ok = (n0 + c0) < N;  // whole 16-chunk in or out
      long boff = offB + (long)(k0 + kk) * ldb + (ok ? n0 + c0 : 0);
      bf16x8 x0 = *(const bf16x8*)(Bm + boff);
      bf16x8 x1 = *(const bf16x8*)(Bm + boff + 8);
      bf16_t zz = (bf16_t)0.0f;
#pragma unroll
      for (int i = 0; i < 8; i++) {
        ldsB[(c0 + i) * LDSK + kk]     = ok ? x0[i] : zz;
        ldsB[(c0 + 8 + i) * LDSK + kk] = ok ? x1[i] : zz;
      }
    }
#if USE_TDM
    if (tid < 32) __builtin_amdgcn_s_wait_tensorcnt(0);  // TDM data landed in LDS
#endif
    __syncthreads();

    // A fragments (ISA 16-bit A 16x32 layout: lanes<16 K=0..7|16..23, lanes>=16 +8)
    bf16x16 af[4];
#pragma unroll
    for (int mt = 0; mt < 4; ++mt) {
      const bf16_t* p = &ldsA[(wm * 64 + mt * 16 + rA) * LDSK];
      bf16x8 lo = *(const bf16x8*)(p + kbA);
      bf16x8 hi = *(const bf16x8*)(p + kbA + 16);
      bf16x16 r;
#pragma unroll
      for (int i = 0; i < 8; i++) { r[i] = lo[i]; r[i + 8] = hi[i]; }
      af[mt] = r;
    }
    // B fragments (lane = column, K chunks at kbB, kbB+16 per half-wave)
    bf16x16 bfm[2];
#pragma unroll
    for (int nt = 0; nt < 2; ++nt) {
      const bf16_t* p = &ldsB[(wn * 32 + nt * 16 + rA) * LDSK + kbB];
      bf16x8 lo = *(const bf16x8*)(p);
      bf16x8 hi = *(const bf16x8*)(p + 16);
      bf16x16 r;
#pragma unroll
      for (int i = 0; i < 8; i++) { r[i] = lo[i]; r[i + 8] = hi[i]; }
      bfm[nt] = r;
    }
#pragma unroll
    for (int mt = 0; mt < 4; ++mt)
#pragma unroll
      for (int nt = 0; nt < 2; ++nt)
        acc[mt][nt] = __builtin_amdgcn_wmma_f32_16x16x32_bf16(
            false, af[mt], false, bfm[nt], (short)0, acc[mt][nt], false, false);
    __syncthreads();
  }

  // epilogue: C layout lane 0-15 -> N=lane, M=r ; lane 16-31 -> N=lane-16, M=r+8
  const int mh = (lane >> 4) * 8;
  const int nl = lane & 15;
#pragma unroll
  for (int mt = 0; mt < 4; ++mt) {
#pragma unroll
    for (int nt = 0; nt < 2; ++nt) {
      int gn = n0 + wn * 32 + nt * 16 + nl;
      if (gn >= N) continue;
      float bv = bias ? bias[gn] : 0.0f;
#pragma unroll
      for (int r = 0; r < 8; r++) {
        int gm = m0 + wm * 64 + mt * 16 + mh + r;
        if (gm >= M) continue;
        float v = alpha * acc[mt][nt][r] + bv;
        long idx = offC + (long)gm * ldc + gn;
        if (EPI == 0) {
          Cf[idx] = v;
        } else if (EPI == 1) {
          Cb[idx] = (bf16_t)v;
        } else if (EPI == 2) {
          Cb[idx] = (bf16_t)gelu_tanh(v);
        } else if (EPI == 3) {
          float s = rs ? rs[(long)gm * rsStride] : 1.0f;
          Cf[idx] += s * v;
        }
      }
    }
  }
}

// ---------------------------- helper kernels -------------------------------

__global__ void f32_to_bf16_k(const float* __restrict__ in, bf16_t* __restrict__ out, long n) {
  long i = (long)blockIdx.x * blockDim.x + threadIdx.x;
  long st = (long)gridDim.x * blockDim.x;
  for (; i < n; i += st) out[i] = (bf16_t)in[i];
}

__global__ __launch_bounds__(256)
void layernorm_bf16_k(const float* __restrict__ x, const float* __restrict__ g,
                      const float* __restrict__ b, bf16_t* __restrict__ y, int Hd) {
  __shared__ float red[256];
  const long row = blockIdx.x;
  const float* xr = x + row * (long)Hd;
  int tid = threadIdx.x;
  float s = 0.f;
  for (int i = tid; i < Hd; i += 256) s += xr[i];
  red[tid] = s; __syncthreads();
  for (int off = 128; off > 0; off >>= 1) { if (tid < off) red[tid] += red[tid + off]; __syncthreads(); }
  float mean = red[0] / Hd;
  __syncthreads();
  float v = 0.f;
  for (int i = tid; i < Hd; i += 256) { float d = xr[i] - mean; v += d * d; }
  red[tid] = v; __syncthreads();
  for (int off = 128; off > 0; off >>= 1) { if (tid < off) red[tid] += red[tid + off]; __syncthreads(); }
  float inv = rsqrtf(red[0] / Hd + 1e-5f);
  for (int i = tid; i < Hd; i += 256)
    y[row * (long)Hd + i] = (bf16_t)((xr[i] - mean) * inv * g[i] + b[i]);
}

template <int OBF>
__global__ __launch_bounds__(256)
void softmax_k(const float* __restrict__ in, float* __restrict__ of,
               bf16_t* __restrict__ ob, int L) {
  __shared__ float red[256];
  const long row = blockIdx.x;
  const float* x = in + row * (long)L;
  int tid = threadIdx.x;
  float m = -3.4e38f;
  for (int i = tid; i < L; i += 256) m = fmaxf(m, x[i]);
  red[tid] = m; __syncthreads();
  for (int off = 128; off > 0; off >>= 1) { if (tid < off) red[tid] = fmaxf(red[tid], red[tid + off]); __syncthreads(); }
  m = red[0]; __syncthreads();
  float s = 0.f;
  for (int i = tid; i < L; i += 256) s += __expf(x[i] - m);
  red[tid] = s; __syncthreads();
  for (int off = 128; off > 0; off >>= 1) { if (tid < off) red[tid] += red[tid + off]; __syncthreads(); }
  float inv = 1.f / red[0];
  for (int i = tid; i < L; i += 256) {
    float v = __expf(x[i] - m) * inv;
    if (OBF) ob[row * (long)L + i] = (bf16_t)v;
    else     of[row * (long)L + i] = v;
  }
}

__global__ void mean_rows_k(const float* __restrict__ x, float* __restrict__ ctx,
                            int S, int Hd, long total) {
  long i = (long)blockIdx.x * blockDim.x + threadIdx.x;
  if (i >= total) return;
  long bq = i / Hd; int h = (int)(i - bq * Hd);
  float s = 0.f;
  for (int t = 0; t < S; t++) s += x[(bq * S + t) * (long)Hd + h];
  ctx[i] = s / (float)S;
}

__global__ void concat_ctx_k(const float* __restrict__ x, const float* __restrict__ ctx,
                             bf16_t* __restrict__ out, int S, int Hd, long total) {
  long i = (long)blockIdx.x * blockDim.x + threadIdx.x;
  long st = (long)gridDim.x * blockDim.x;
  for (; i < total; i += st) {
    long bs = i / (2 * Hd); int c = (int)(i - bs * (2 * Hd));
    long bq = bs / S;
    float v = (c < Hd) ? x[bs * (long)Hd + c] : ctx[bq * (long)Hd + (c - Hd)];
    out[i] = (bf16_t)v;
  }
}

__global__ void zero_f32_k(float* __restrict__ p, long n) {
  long i = (long)blockIdx.x * blockDim.x + threadIdx.x;
  long st = (long)gridDim.x * blockDim.x;
  for (; i < n; i += st) p[i] = 0.f;
}

// expert-choice top-k per (batch, expert): one block per (b,e)
__global__ __launch_bounds__(256)
void topk_mask_k(const float* __restrict__ probs, float* __restrict__ mask,
                 int S, int E, int kk) {
  int b = blockIdx.x, e = blockIdx.y;
  __shared__ float vals[640];
  __shared__ float rv[256];
  __shared__ int   ri[256];
  int tid = threadIdx.x;
  for (int s = tid; s < S; s += 256) {
    vals[s] = probs[((long)b * S + s) * E + e];
    mask[((long)b * S + s) * E + e] = 0.f;
  }
  __syncthreads();
  for (int it = 0; it < kk; ++it) {
    float best = -3.4e38f; int bi = S;
    for (int s = tid; s < S; s += 256) {
      float v = vals[s];
      if (v > best || (v == best && s < bi)) { best = v; bi = s; }
    }
    rv[tid] = best; ri[tid] = bi;
    __syncthreads();
    for (int off = 128; off > 0; off >>= 1) {
      if (tid < off) {
        float ov = rv[tid + off]; int oi = ri[tid + off];
        if (ov > rv[tid] || (ov == rv[tid] && oi < ri[tid])) { rv[tid] = ov; ri[tid] = oi; }
      }
      __syncthreads();
    }
    if (tid == 0) {
      int s = ri[0];
      if (s < S) { mask[((long)b * S + s) * E + e] = 1.f; vals[s] = -3.4e38f; }
    }
    __syncthreads();
  }
}

__global__ void moe_combine_k(const float* __restrict__ res, const float* __restrict__ acc,
                              const float* __restrict__ mask, float* __restrict__ out,
                              int Hd, int E, long total) {
  long i = (long)blockIdx.x * blockDim.x + threadIdx.x;
  long st = (long)gridDim.x * blockDim.x;
  for (; i < total; i += st) {
    long bs = i / Hd;
    float c = 0.f;
    for (int e = 0; e < E; e++) c += mask[bs * E + e];
    c = fmaxf(c, 1.f);
    out[i] = res[i] + acc[i] / c;
  }
}

// ------------------------------ host driver --------------------------------

static inline void gemm_launch(int epi, int opb,
                               const bf16_t* A, const bf16_t* Bm, const float* bias,
                               float* Cf, bf16_t* Cb,
                               int M, int N, int K, int lda, int ldb, int ldc,
                               long sAb, long sAh, long sBb, long sBh, long sCb, long sCh,
                               int nh, int batch, float alpha,
                               const float* rs, int rsStride, hipStream_t st) {
  dim3 g((N + BN - 1) / BN, (M + BM - 1) / BM, batch), blk(256);
#define GL(E_, O_) gemm_bf16_k<E_, O_><<<g, blk, 0, st>>>( \
    A, Bm, bias, Cf, Cb, M, N, K, lda, ldb, ldc, sAb, sAh, sBb, sBh, sCb, sCh, nh, alpha, rs, rsStride)
  if      (epi == 0 && opb == 0) GL(0, 0);
  else if (epi == 1 && opb == 0) GL(1, 0);
  else if (epi == 1 && opb == 1) GL(1, 1);
  else if (epi == 2 && opb == 0) GL(2, 0);
  else if (epi == 3 && opb == 0) GL(3, 0);
#undef GL
}

extern "C" void kernel_launch(void* const* d_in, const int* in_sizes, int n_in,
                              void* d_out, int out_size, void* d_ws, size_t ws_size,
                              hipStream_t stream) {
  (void)in_sizes; (void)n_in; (void)out_size; (void)ws_size;
  constexpr int B = 4, T = 512, V = 576, Lc = 256, H = 1024, NH = 16, I = 4096, E = 8, HD = 64;

  const float* query_tokens = (const float*)d_in[0];
  const float* image_tokens = (const float*)d_in[1];
  const float* text_context = (const float*)d_in[2];
  const float* sa_w_in  = (const float*)d_in[3];  const float* sa_b_in  = (const float*)d_in[4];
  const float* sa_w_out = (const float*)d_in[5];  const float* sa_b_out = (const float*)d_in[6];
  const float* ca_w_in  = (const float*)d_in[7];  const float* ca_b_in  = (const float*)d_in[8];
  const float* ca_w_out = (const float*)d_in[9];  const float* ca_b_out = (const float*)d_in[10];
  const float* gate_img_w = (const float*)d_in[11]; const float* gate_img_b = (const float*)d_in[12];
  const float* gate_txt_w = (const float*)d_in[13]; const float* gate_txt_b = (const float*)d_in[14];
  const float* e_w1 = (const float*)d_in[15]; const float* e_b1 = (const float*)d_in[16];
  const float* e_w2 = (const float*)d_in[17]; const float* e_b2 = (const float*)d_in[18];
  const float* lnq_g = (const float*)d_in[19]; const float* lnq_b = (const float*)d_in[20];
  const float* lnc_g = (const float*)d_in[21]; const float* lnc_b = (const float*)d_in[22];
  const float* lnf_g = (const float*)d_in[23]; const float* lnf_b = (const float*)d_in[24];

  // ---- workspace carve ----
  char* wsp = (char*)d_ws;
  auto take = [&](size_t bytes) -> char* {
    char* p = wsp; wsp += (bytes + 255) & ~(size_t)255; return p;
  };
  bf16_t* w_sa_in  = (bf16_t*)take((size_t)3 * H * H * 2);
  bf16_t* w_sa_out = (bf16_t*)take((size_t)H * H * 2);
  bf16_t* w_ca_in  = (bf16_t*)take((size_t)3 * H * H * 2);
  bf16_t* w_ca_out = (bf16_t*)take((size_t)H * H * 2);
  bf16_t* w_gi     = (bf16_t*)take((size_t)E * 2 * H * 2);
  bf16_t* w_gt     = (bf16_t*)take((size_t)E * 2 * H * 2);
  bf16_t* w_e1b    = (bf16_t*)take((size_t)E * I * H * 2);
  bf16_t* w_e2b    = (bf16_t*)take((size_t)E * H * I * 2);
  float*  qres     = (float*) take((size_t)B * T * H * 4);
  bf16_t* xn       = (bf16_t*)take((size_t)B * T * H * 2);
  bf16_t* qb       = (bf16_t*)take((size_t)B * T * H * 2);
  bf16_t* kb       = (bf16_t*)take((size_t)B * V * H * 2);
  bf16_t* vb       = (bf16_t*)take((size_t)B * V * H * 2);
  bf16_t* ob       = (bf16_t*)take((size_t)B * T * H * 2);
  bf16_t* imgb     = (bf16_t*)take((size_t)B * V * H * 2);
  float*  scores   = (float*) take((size_t)B * NH * T * V * 4);
  bf16_t* Pb       = (bf16_t*)take((size_t)B * NH * T * V * 2);
  bf16_t* gate_in  = (bf16_t*)take((size_t)B * V * 2 * H * 2);
  float*  logits   = (float*) take((size_t)B * V * E * 4);
  float*  probs    = (float*) take((size_t)B * V * E * 4);
  float*  mask_img = (float*) take((size_t)B * V * E * 4);
  float*  mask_txt = (float*) take((size_t)B * T * E * 4);
  float*  ctx_img  = (float*) take((size_t)B * H * 4);
  float*  ctx_txt  = (float*) take((size_t)B * H * 4);
  bf16_t* hbuf     = (bf16_t*)take((size_t)B * V * I * 2);
  float*  accT     = (float*) take((size_t)B * T * H * 4);
  float*  accV     = (float*) take((size_t)B * V * H * 4);

  float* out_q   = (float*)d_out;
  float* out_img = (float*)d_out + (long)B * T * H;

  auto cvt = [&](const float* src, bf16_t* dst, long n) {
    long bl = (n + 255) / 256; if (bl > 4096) bl = 4096;
    f32_to_bf16_k<<<dim3((unsigned)bl), 256, 0, stream>>>(src, dst, n);
  };
  auto gsz = [](long n) { long b = (n + 255) / 256; if (b > 262144) b = 262144; return dim3((unsigned)b); };

  // ---- weight conversion fp32 -> bf16 ----
  cvt(sa_w_in,  w_sa_in,  (long)3 * H * H);
  cvt(sa_w_out, w_sa_out, (long)H * H);
  cvt(ca_w_in,  w_ca_in,  (long)3 * H * H);
  cvt(ca_w_out, w_ca_out, (long)H * H);
  cvt(gate_img_w, w_gi, (long)E * 2 * H);
  cvt(gate_txt_w, w_gt, (long)E * 2 * H);
  cvt(e_w1, w_e1b, (long)E * I * H);
  cvt(e_w2, w_e2b, (long)E * H * I);
  cvt(image_tokens, imgb, (long)B * V * H);

  (void)hipMemcpyAsync(qres, query_tokens, (size_t)B * T * H * 4, hipMemcpyDeviceToDevice, stream);

  // ================= self-attention =================
  layernorm_bf16_k<<<B * T, 256, 0, stream>>>(qres, lnq_g, lnq_b, xn, H);
  gemm_launch(1, 0, xn, w_sa_in,                 sa_b_in,         nullptr, qb, B * T, H, H, H, H, H,
              0, 0, 0, 0, 0, 0, 1, 1, 1.f, nullptr, 0, stream);
  gemm_launch(1, 0, xn, w_sa_in + (long)H * H,   sa_b_in + H,     nullptr, kb, B * T, H, H, H, H, H,
              0, 0, 0, 0, 0, 0, 1, 1, 1.f, nullptr, 0, stream);
  gemm_launch(1, 0, xn, w_sa_in + 2L * H * H,    sa_b_in + 2 * H, nullptr, vb, B * T, H, H, H, H, H,
              0, 0, 0, 0, 0, 0, 1, 1, 1.f, nullptr, 0, stream);
  // scores[b,h,q,k] = q . k / sqrt(HD)
  gemm_launch(0, 0, qb, kb, nullptr, scores, nullptr, T, T, HD, H, H, T,
              (long)T * H, HD, (long)T * H, HD, (long)NH * T * T, (long)T * T,
              NH, B * NH, 0.125f, nullptr, 0, stream);
  softmax_k<1><<<B * NH * T, 256, 0, stream>>>(scores, nullptr, Pb, T);
  // O = P @ V
  gemm_launch(1, 1, Pb, vb, nullptr, nullptr, ob, T, HD, T, T, H, H,
              (long)NH * T * T, (long)T * T, (long)T * H, HD, (long)T * H, HD,
              NH, B * NH, 1.f, nullptr, 0, stream);
  // qres += O @ sa_w_out^T + b
  gemm_launch(3, 0, ob, w_sa_out, sa_b_out, qres, nullptr, B * T, H, H, H, H, H,
              0, 0, 0, 0, 0, 0, 1, 1, 1.f, nullptr, 0, stream);

  // ================= cross-attention =================
  layernorm_bf16_k<<<B * T, 256, 0, stream>>>(qres, lnc_g, lnc_b, xn, H);
  gemm_launch(1, 0, xn,   w_ca_in,               ca_b_in,         nullptr, qb, B * T, H, H, H, H, H,
              0, 0, 0, 0, 0, 0, 1, 1, 1.f, nullptr, 0, stream);
  gemm_launch(1, 0, imgb, w_ca_in + (long)H * H, ca_b_in + H,     nullptr, kb, B * V, H, H, H, H, H,
              0, 0, 0, 0, 0, 0, 1, 1, 1.f, nullptr, 0, stream);
  gemm_launch(1, 0, imgb, w_ca_in + 2L * H * H,  ca_b_in + 2 * H, nullptr, vb, B * V, H, H, H, H, H,
              0, 0, 0, 0, 0, 0, 1, 1, 1.f, nullptr, 0, stream);
  gemm_launch(0, 0, qb, kb, nullptr, scores, nullptr, T, V, HD, H, H, V,
              (long)T * H, HD, (long)V * H, HD, (long)NH * T * V, (long)T * V,
              NH, B * NH, 0.125f, nullptr, 0, stream);
  softmax_k<1><<<B * NH * T, 256, 0, stream>>>(scores, nullptr, Pb, V);
  gemm_launch(1, 1, Pb, vb, nullptr, nullptr, ob, T, HD, V, V, H, H,
              (long)NH * T * V, (long)T * V, (long)V * H, HD, (long)T * H, HD,
              NH, B * NH, 1.f, nullptr, 0, stream);
  gemm_launch(3, 0, ob, w_ca_out, ca_b_out, qres, nullptr, B * T, H, H, H, H, H,
              0, 0, 0, 0, 0, 0, 1, 1, 1.f, nullptr, 0, stream);

  // ================= gating =================
  mean_rows_k<<<(B * H + 255) / 256, 256, 0, stream>>>(image_tokens, ctx_img, V, H, (long)B * H);
  mean_rows_k<<<(B * H + 255) / 256, 256, 0, stream>>>(text_context, ctx_txt, Lc, H, (long)B * H);

  int capT = ((int)(1.25f * T) + E - 1) / E; if (capT < 1) capT = 1; if (capT > T) capT = T;   // 80
  int capV = ((int)(1.25f * V) + E - 1) / E; if (capV < 1) capV = 1; if (capV > V) capV = V;   // 90

  // image stream gating: [image_tokens | text_ctx]
  concat_ctx_k<<<gsz((long)B * V * 2 * H), 256, 0, stream>>>(image_tokens, ctx_txt, gate_in, V, H, (long)B * V * 2 * H);
  gemm_launch(0, 0, gate_in, w_gi, gate_img_b, logits, nullptr, B * V, E, 2 * H, 2 * H, 2 * H, E,
              0, 0, 0, 0, 0, 0, 1, 1, 1.f, nullptr, 0, stream);
  softmax_k<0><<<B * V, 256, 0, stream>>>(logits, probs, nullptr, E);
  topk_mask_k<<<dim3(B, E), 256, 0, stream>>>(probs, mask_img, V, E, capV);

  // text stream gating: [q | image_ctx]
  concat_ctx_k<<<gsz((long)B * T * 2 * H), 256, 0, stream>>>(qres, ctx_img, gate_in, T, H, (long)B * T * 2 * H);
  gemm_launch(0, 0, gate_in, w_gt, gate_txt_b, logits, nullptr, B * T, E, 2 * H, 2 * H, 2 * H, E,
              0, 0, 0, 0, 0, 0, 1, 1, 1.f, nullptr, 0, stream);
  softmax_k<0><<<B * T, 256, 0, stream>>>(logits, probs, nullptr, E);
  topk_mask_k<<<dim3(B, E), 256, 0, stream>>>(probs, mask_txt, T, E, capT);

  // ================= MoE (text stream) =================
  layernorm_bf16_k<<<B * T, 256, 0, stream>>>(qres, lnf_g, lnf_b, xn, H);
  zero_f32_k<<<gsz((long)B * T * H), 256, 0, stream>>>(accT, (long)B * T * H);
  for (int e = 0; e < E; ++e) {
    gemm_launch(2, 0, xn, w_e1b + (long)e * I * H, e_b1 + (long)e * I, nullptr, hbuf,
                B * T, I, H, H, H, I, 0, 0, 0, 0, 0, 0, 1, 1, 1.f, nullptr, 0, stream);
    gemm_launch(3, 0, hbuf, w_e2b + (long)e * H * I, e_b2 + (long)e * H, accT, nullptr,
                B * T, H, I, I, I, H, 0, 0, 0, 0, 0, 0, 1, 1, 1.f, mask_txt + e, E, stream);
  }
  moe_combine_k<<<gsz((long)B * T * H), 256, 0, stream>>>(qres, accT, mask_txt, out_q, H, E, (long)B * T * H);

  // ================= MoE (image stream) =================
  zero_f32_k<<<gsz((long)B * V * H), 256, 0, stream>>>(accV, (long)B * V * H);
  for (int e = 0; e < E; ++e) {
    gemm_launch(2, 0, imgb, w_e1b + (long)e * I * H, e_b1 + (long)e * I, nullptr, hbuf,
                B * V, I, H, H, H, I, 0, 0, 0, 0, 0, 0, 1, 1, 1.f, nullptr, 0, stream);
    gemm_launch(3, 0, hbuf, w_e2b + (long)e * H * I, e_b2 + (long)e * H, accV, nullptr,
                B * V, H, I, I, I, H, 0, 0, 0, 0, 0, 0, 1, 1, 1.f, mask_img + e, E, stream);
  }
  moe_combine_k<<<gsz((long)B * V * H), 256, 0, stream>>>(image_tokens, accV, mask_img, out_img, H, E, (long)B * V * H);
}
